// MolGIN_19095424598470
// MI455X (gfx1250) — matrix operand order repaired and checked by reading, not compiled, not run on hardware
//
#include <hip/hip_runtime.h>

// ---------------------------------------------------------------------------
// MolGIN forward for MI455X (gfx1250, wave32, WMMA).
//   - CSR built once per call (edge_index is layer-invariant) so the per-layer
//     neighbor aggregation is a pure gather (no float atomics; h fits in L2).
//   - GEMMs: 128x128 block tiles, v_wmma_f32_16x16x32_f16 with f32 accum,
//     fused per-column sum/sumsq (ds_add_f32 -> global atomics) for BatchNorm.
//   - GEMM2's f16 A operand staged via GLOBAL_LOAD_ASYNC_TO_LDS_B128 when the
//     toolchain exposes the builtin (ASYNCcnt path), else VGPR staging.
// ---------------------------------------------------------------------------

typedef __attribute__((ext_vector_type(16))) _Float16 v16h;
typedef __attribute__((ext_vector_type(8)))  _Float16 v8h;
typedef __attribute__((ext_vector_type(8)))  float    v8f;
typedef int v4i __attribute__((vector_size(16)));   // matches builtin param

#define DEV __device__ __forceinline__

#if __has_builtin(__builtin_amdgcn_global_load_async_to_lds_b128)
#define HAVE_ASYNC_LDS 1
#endif

DEV float atomAddGlobal(float* p, float v) {
  return __hip_atomic_fetch_add(p, v, __ATOMIC_RELAXED, __HIP_MEMORY_SCOPE_AGENT);
}
DEV float atomAddShared(float* p, float v) {
  return __hip_atomic_fetch_add(p, v, __ATOMIC_RELAXED, __HIP_MEMORY_SCOPE_WORKGROUP);
}
DEV void waitAsync0() {
#ifdef HAVE_ASYNC_LDS
#if __has_builtin(__builtin_amdgcn_s_wait_asynccnt)
  __builtin_amdgcn_s_wait_asynccnt(0);
#else
  asm volatile("s_wait_asynccnt 0x0" ::: "memory");
#endif
#endif
}

// ---------------------------------------------------------------------------
__global__ void zero_f32_kernel(float* __restrict__ p, long n) {
  long i = (long)blockIdx.x * blockDim.x + threadIdx.x;
  if (i < n) p[i] = 0.f;
}

// ------------------------------- CSR build ---------------------------------
__global__ void count_kernel(const int* __restrict__ ei, int* __restrict__ cnt,
                             int E) {
  const int e = blockIdx.x * blockDim.x + threadIdx.x;
  if (e < E) atomicAdd(&cnt[ei[E + e]], 1);
}

// Block scans 1024 elements (256 thr x 4); exclusive scan out + block total.
__global__ void scan1_kernel(const int* __restrict__ in, int* __restrict__ out,
                             int* __restrict__ blksum, int n) {
  __shared__ int sh[256];
  const int t = threadIdx.x;
  const int base = blockIdx.x * 1024;
  int v[4];
  int s = 0;
#pragma unroll
  for (int q = 0; q < 4; ++q) {
    const int idx = base + t * 4 + q;
    v[q] = (idx < n) ? in[idx] : 0;
    s += v[q];
  }
  sh[t] = s;
  __syncthreads();
  for (int off = 1; off < 256; off <<= 1) {
    const int x = (t >= off) ? sh[t - off] : 0;
    __syncthreads();
    sh[t] += x;
    __syncthreads();
  }
  if (t == 255) blksum[blockIdx.x] = sh[255];
  int run = sh[t] - s;  // exclusive prefix of this thread's chunk
#pragma unroll
  for (int q = 0; q < 4; ++q) {
    const int idx = base + t * 4 + q;
    if (idx < n) out[idx] = run;
    run += v[q];
  }
}

__global__ void scan_spine_kernel(int* __restrict__ blksum, int nb) {
  if (threadIdx.x == 0) {
    int acc = 0;
    for (int i = 0; i < nb; ++i) {
      const int x = blksum[i];
      blksum[i] = acc;
      acc += x;
    }
  }
}

__global__ void scan_add_kernel(int* __restrict__ out,
                                const int* __restrict__ blksum, int n, int E) {
  const int i = blockIdx.x * blockDim.x + threadIdx.x;
  if (i < n) out[i] += blksum[i >> 10];
  if (i == 0) out[n] = E;
}

__global__ void fill_kernel(const int* __restrict__ ei,
                            const int* __restrict__ rowptr,
                            int* __restrict__ cursor, int* __restrict__ srcidx,
                            int E) {
  const int e = blockIdx.x * blockDim.x + threadIdx.x;
  if (e >= E) return;
  const int d = ei[E + e];
  const int pos = atomicAdd(&cursor[d], 1);
  srcidx[rowptr[d] + pos] = ei[e];
}

// ---------------------------------------------------------------------------
// AtomEncoder: h[n][c] = sum_j emb[x[n][j] + off[j]][c]; one block per node.
__global__ void atom_encoder_kernel(const int* __restrict__ x,
                                    const float* __restrict__ emb,
                                    float* __restrict__ h, int H) {
  __shared__ int rows[9];
  static const int off[9] = {0, 119, 129, 140, 152, 161, 166, 174, 176};
  const int n = blockIdx.x;
  const int c = threadIdx.x;
  if (c < 9) rows[c] = x[n * 9 + c] + off[c];
  __syncthreads();
  float s = 0.f;
#pragma unroll
  for (int j = 0; j < 9; ++j) s += emb[(long)rows[j] * H + c];
  h[(long)n * H + c] = s;
}

// z[n] = (1+eps)*h[n] + sum_{s in in(n)} h[s].  One wave per node, f4/lane.
__global__ void gather_z_kernel(const float* __restrict__ h,
                                const int* __restrict__ rowptr,
                                const int* __restrict__ srcidx,
                                const float* __restrict__ eps_l,
                                float* __restrict__ z, int N) {
  const int n = (blockIdx.x * blockDim.x + threadIdx.x) >> 5;
  const int lane = threadIdx.x & 31;
  if (n >= N) return;
  const int beg = rowptr[n];
  const int end = rowptr[n + 1];
  const float epv = 1.f + eps_l[0];
  float4 acc = *(const float4*)(h + (long)n * 128 + lane * 4);
  acc.x *= epv; acc.y *= epv; acc.z *= epv; acc.w *= epv;
  for (int j = beg; j < end; ++j) {
    const int s = srcidx[j];  // wave-uniform -> scalar load
    const float4 v = *(const float4*)(h + (long)s * 128 + lane * 4);
    acc.x += v.x; acc.y += v.y; acc.z += v.z; acc.w += v.w;
  }
  *(float4*)(z + (long)n * 128 + lane * 4) = acc;
}

// ---------------------------------------------------------------------------
// Tiled WMMA GEMM: Out[M,NT] = A[M,K] @ W[K,NT] + bias, f32 accumulate.
// Block = 256 threads = 8 waves, 128x128 output tile, K chunked by 64 in LDS.
// ---------------------------------------------------------------------------
#define GM 128
#define GN 128
#define KC 64
#define ASTR 72   // padded LDS row stride (halfs) to spread banks
#define WSTR 72

template <bool A_F16, bool OUT_F16, bool STATS>
__global__ void __launch_bounds__(256)
gemm_kernel(const void* __restrict__ Av, const float* __restrict__ W,
            const float* __restrict__ bias, void* __restrict__ Outv,
            float* __restrict__ stat_sum, float* __restrict__ stat_sq,
            int M, int K, int NT) {
  __shared__ __attribute__((aligned(16))) _Float16 Ash[GM * ASTR];
  __shared__ __attribute__((aligned(16))) _Float16 Wsh[GN * WSTR];
  __shared__ float colsum[GN];
  __shared__ float colsq[GN];

  const int tid = threadIdx.x;
  const int m_base = blockIdx.x * GM;
  const int nb = blockIdx.y * GN;

  if (STATS && tid < GN) { colsum[tid] = 0.f; colsq[tid] = 0.f; }

  const int wv = tid >> 5;
  const int lane = tid & 31;
  const int hlf = lane >> 4;   // lanes 0-15 vs 16-31
  const int l16 = lane & 15;

  v8f acc[8];
#pragma unroll
  for (int t = 0; t < 8; ++t)
#pragma unroll
    for (int r = 0; r < 8; ++r) acc[t][r] = 0.f;

  for (int k0 = 0; k0 < K; k0 += KC) {
    // ---- stage A tile (GM x KC) as f16, zero-pad rows >= M ----
    if (A_F16) {
      const _Float16* A = (const _Float16*)Av;
      for (int i = tid * 8; i < GM * KC; i += 256 * 8) {
        const int row = i >> 6;
        const int kk = i & 63;
        const int rg = m_base + row;
        if (rg < M) {
#ifdef HAVE_ASYNC_LDS
          // Direct global->LDS async copy (ASYNCcnt), no VGPR roundtrip.
          __builtin_amdgcn_global_load_async_to_lds_b128(
              (__attribute__((address_space(1))) v4i*)(A + (long)rg * K + k0 + kk),
              (__attribute__((address_space(3))) v4i*)(Ash + row * ASTR + kk),
              0, 0);
#else
          *(v8h*)(Ash + row * ASTR + kk) =
              *(const v8h*)(A + (long)rg * K + k0 + kk);
#endif
        } else {
          v8h zv;
#pragma unroll
          for (int q = 0; q < 8; ++q) zv[q] = (_Float16)0.f;
          *(v8h*)(Ash + row * ASTR + kk) = zv;
        }
      }
      waitAsync0();
    } else {
      const float* A = (const float*)Av;
      for (int i = tid * 4; i < GM * KC; i += 256 * 4) {
        const int row = i >> 6;
        const int kk = i & 63;
        const int rg = m_base + row;
        float4 v = make_float4(0.f, 0.f, 0.f, 0.f);
        if (rg < M) v = *(const float4*)(A + (long)rg * K + k0 + kk);
        _Float16* d = Ash + row * ASTR + kk;
        d[0] = (_Float16)v.x; d[1] = (_Float16)v.y;
        d[2] = (_Float16)v.z; d[3] = (_Float16)v.w;
      }
    }
    // ---- stage W chunk transposed: Wsh[c][kk] = W[k0+kk][nb+c] ----
    for (int i = tid; i < GN * KC; i += 256) {
      const int kk = i >> 7;
      const int c = i & 127;
      Wsh[c * WSTR + kk] = (_Float16)W[(long)(k0 + kk) * NT + nb + c];
    }
    __syncthreads();

    // ---- compute: wave wv owns row-tile wv, sweeps 8 col-tiles ----
    const int arow = (wv * 16 + l16) * ASTR;
#pragma unroll
    for (int ct = 0; ct < 8; ++ct) {
#pragma unroll
      for (int ks = 0; ks < KC; ks += 32) {
        // A frag: lanes<16 -> K[0..7],K[16..23]; lanes>=16 -> K[8..15],K[24..31]
        v8h alo = *(const v8h*)(Ash + arow + ks + hlf * 8);
        v8h ahi = *(const v8h*)(Ash + arow + ks + 16 + hlf * 8);
        v16h a = __builtin_shufflevector(alo, ahi, 0, 1, 2, 3, 4, 5, 6, 7, 8,
                                         9, 10, 11, 12, 13, 14, 15);
        // B frag: lanes<16 -> K[0..7],K[16..23] of col l16; lanes>=16 -> K[8..15],K[24..31]
        const _Float16* bp = Wsh + (ct * 16 + l16) * WSTR + ks + hlf * 8;
        v8h blo = *(const v8h*)(bp);
        v8h bhi = *(const v8h*)(bp + 16);
        v16h b = __builtin_shufflevector(blo, bhi, 0, 1, 2, 3, 4, 5, 6, 7, 8,
                                         9, 10, 11, 12, 13, 14, 15);
        acc[ct] = __builtin_amdgcn_wmma_f32_16x16x32_f16(
            false, a, false, b, (short)0, acc[ct], false, false);
      }
    }
    __syncthreads();
  }

  // ---- epilogue: bias, store, fused column stats ----
  const int row0 = m_base + wv * 16 + hlf * 8;   // VGPR r -> rows r / r+8
#pragma unroll
  for (int ct = 0; ct < 8; ++ct) {
    const int col_l = ct * 16 + l16;
    const int col_g = nb + col_l;
    const float bv = bias[col_g];
    float s = 0.f, s2 = 0.f;
#pragma unroll
    for (int r = 0; r < 8; ++r) {
      const int rg = row0 + r;
      if (rg < M) {
        const float v = acc[ct][r] + bv;
        if (OUT_F16)
          ((_Float16*)Outv)[(long)rg * NT + col_g] = (_Float16)v;
        else
          ((float*)Outv)[(long)rg * NT + col_g] = v;
        s += v;
        s2 += v * v;
      }
    }
    if (STATS) {
      atomAddShared(&colsum[col_l], s);   // ds_add_f32
      atomAddShared(&colsq[col_l], s2);
    }
  }
  if (STATS) {
    __syncthreads();
    if (tid < GN) {
      atomAddGlobal(&stat_sum[nb + tid], colsum[tid]);
      atomAddGlobal(&stat_sq[nb + tid], colsq[tid]);
    }
  }
}

// scale = gamma*rsqrt(var+eps); shift = beta - scale*mean (biased var, as ref)
__global__ void bn_finalize_kernel(const float* __restrict__ sum,
                                   const float* __restrict__ sq,
                                   const float* __restrict__ gamma,
                                   const float* __restrict__ beta,
                                   float* __restrict__ scale,
                                   float* __restrict__ shift, int C, float invN) {
  const int c = blockIdx.x * blockDim.x + threadIdx.x;
  if (c >= C) return;
  const float m = sum[c] * invN;
  const float var = sq[c] * invN - m * m;
  const float sc = gamma[c] * rsqrtf(var + 1e-5f);
  scale[c] = sc;
  shift[c] = beta[c] - sc * m;
}

// t = relu(scale*t + shift), in place on f16 (feeds next GEMM directly)
__global__ void bn_apply_f16_kernel(_Float16* __restrict__ t,
                                    const float* __restrict__ scale,
                                    const float* __restrict__ shift, long n,
                                    int cmask) {
  long i = (long)blockIdx.x * blockDim.x + threadIdx.x;
  if (i >= n) return;
  const int c = (int)(i & cmask);
  const float v = (float)t[i] * scale[c] + shift[c];
  t[i] = (_Float16)fmaxf(v, 0.f);
}

// h = relu(scale*t2 + shift), f32 (h stays f32 for next layer's gather)
__global__ void bn_relu_f32_kernel(const float* __restrict__ in,
                                   float* __restrict__ out,
                                   const float* __restrict__ scale,
                                   const float* __restrict__ shift, long n,
                                   int cmask) {
  long i = (long)blockIdx.x * blockDim.x + threadIdx.x;
  if (i >= n) return;
  const int c = (int)(i & cmask);
  out[i] = fmaxf(in[i] * scale[c] + shift[c], 0.f);
}

// global_add_pool: g[batch_idx[n]] += h[n]   (H = 128)
__global__ void pool_kernel(const float* __restrict__ h,
                            const int* __restrict__ bidx,
                            float* __restrict__ g, long total) {
  long i = (long)blockIdx.x * blockDim.x + threadIdx.x;
  if (i >= total) return;
  const long n = i >> 7;
  const int c = (int)(i & 127);
  atomAddGlobal(&g[(long)bidx[n] * 128 + c], h[i]);
}

// ---------------------------------------------------------------------------
extern "C" void kernel_launch(void* const* d_in, const int* in_sizes, int n_in,
                              void* d_out, int out_size, void* d_ws,
                              size_t ws_size, hipStream_t stream) {
  const int* x = (const int*)d_in[0];
  const int* ei = (const int*)d_in[1];
  const int* bidx = (const int*)d_in[2];
  const float* emb = (const float*)d_in[4];
  const float* W1 = (const float*)d_in[5];
  const float* b1 = (const float*)d_in[6];
  const float* g1 = (const float*)d_in[7];
  const float* be1 = (const float*)d_in[8];
  const float* W2 = (const float*)d_in[9];
  const float* b2 = (const float*)d_in[10];
  const float* g2 = (const float*)d_in[11];
  const float* be2 = (const float*)d_in[12];
  const float* eps = (const float*)d_in[13];
  const float* projW = (const float*)d_in[14];
  const float* projb = (const float*)d_in[15];
  float* out = (float*)d_out;

  const int N = in_sizes[0] / 9;
  const int E = in_sizes[1] / 2;
  const int H = in_sizes[4] / 178;  // 128
  const int H2 = 2 * H;
  const int L = in_sizes[13];
  const int DOUT = in_sizes[14] / H;
  const int G = out_size / DOUT;

  // workspace carve-up (256B aligned)
  char* ws = (char*)d_ws;
  size_t off = 0;
  auto carve = [&](size_t bytes) {
    void* p = ws + off;
    off = (off + bytes + 255) & ~(size_t)255;
    return p;
  };
  float* h = (float*)carve((size_t)N * H * 4);
  float* z = (float*)carve((size_t)N * H * 4);  // also t2 pre-BN
  _Float16* t1 = (_Float16*)carve((size_t)N * H2 * 2);
  float* gp = (float*)carve((size_t)G * H * 4);
  float* ssum = (float*)carve((size_t)H2 * 4);
  float* ssq = (float*)carve((size_t)H2 * 4);
  float* scale = (float*)carve((size_t)H2 * 4);
  float* shift = (float*)carve((size_t)H2 * 4);
  int* rowptr = (int*)carve((size_t)(N + 1) * 4);
  int* cnt = (int*)carve((size_t)N * 4);       // counts, then fill cursor
  int* blks = (int*)carve((size_t)4096 * 4);
  int* srcidx = (int*)carve((size_t)E * 4);
  (void)ws_size;
  (void)n_in;

  const long nH = (long)N * H;
  const long nH2 = (long)N * H2;
  const float invN = 1.f / (float)N;
  const int nbScan = (N + 1023) / 1024;

  // ---- CSR build (once; reused by all layers) ----
  zero_f32_kernel<<<(unsigned)((N + 255) / 256), 256, 0, stream>>>((float*)cnt, N);
  count_kernel<<<(unsigned)((E + 255) / 256), 256, 0, stream>>>(ei, cnt, E);
  scan1_kernel<<<nbScan, 256, 0, stream>>>(cnt, rowptr, blks, N);
  scan_spine_kernel<<<1, 32, 0, stream>>>(blks, nbScan);
  scan_add_kernel<<<(unsigned)((N + 255) / 256), 256, 0, stream>>>(rowptr, blks,
                                                                   N, E);
  zero_f32_kernel<<<(unsigned)((N + 255) / 256), 256, 0, stream>>>((float*)cnt, N);
  fill_kernel<<<(unsigned)((E + 255) / 256), 256, 0, stream>>>(ei, rowptr, cnt,
                                                               srcidx, E);

  atom_encoder_kernel<<<N, H, 0, stream>>>(x, emb, h, H);

  for (int l = 0; l < L; ++l) {
    // z = (1+eps)*h + neighbor-sum  (pure gather, h resident in L2)
    gather_z_kernel<<<(unsigned)((N + 7) / 8), 256, 0, stream>>>(
        h, rowptr, srcidx, eps + l, z, N);

    // GEMM1: t1 = z @ W1_l + b1_l  (f16 out + stats)
    zero_f32_kernel<<<1, 256, 0, stream>>>(ssum, H2);
    zero_f32_kernel<<<1, 256, 0, stream>>>(ssq, H2);
    dim3 grid1((N + GM - 1) / GM, H2 / GN);
    gemm_kernel<false, true, true><<<grid1, 256, 0, stream>>>(
        z, W1 + (size_t)l * H * H2, b1 + (size_t)l * H2, t1, ssum, ssq, N, H,
        H2);
    bn_finalize_kernel<<<(H2 + 255) / 256, 256, 0, stream>>>(
        ssum, ssq, g1 + (size_t)l * H2, be1 + (size_t)l * H2, scale, shift, H2,
        invN);
    bn_apply_f16_kernel<<<(unsigned)((nH2 + 255) / 256), 256, 0, stream>>>(
        t1, scale, shift, nH2, H2 - 1);

    // GEMM2: t2 = t1 @ W2_l + b2_l  (f32 out into z buffer + stats)
    zero_f32_kernel<<<1, 256, 0, stream>>>(ssum, H2);
    zero_f32_kernel<<<1, 256, 0, stream>>>(ssq, H2);
    dim3 grid2((N + GM - 1) / GM, H / GN);
    gemm_kernel<true, false, true><<<grid2, 256, 0, stream>>>(
        t1, W2 + (size_t)l * H2 * H, b2 + (size_t)l * H, z, ssum, ssq, N, H2,
        H);
    bn_finalize_kernel<<<(H + 255) / 256, 256, 0, stream>>>(
        ssum, ssq, g2 + (size_t)l * H, be2 + (size_t)l * H, scale, shift, H,
        invN);
    bn_relu_f32_kernel<<<(unsigned)((nH + 255) / 256), 256, 0, stream>>>(
        z, h, scale, shift, nH, H - 1);
  }

  // global_add_pool + projection
  const long gTot = (long)G * H;
  zero_f32_kernel<<<(unsigned)((gTot + 255) / 256), 256, 0, stream>>>(gp, gTot);
  pool_kernel<<<(unsigned)((nH + 255) / 256), 256, 0, stream>>>(h, bidx, gp, nH);
  dim3 grid3((G + GM - 1) / GM, DOUT / GN);
  gemm_kernel<false, false, false><<<grid3, 256, 0, stream>>>(
      gp, projW, projb, out, nullptr, nullptr, G, H, DOUT);
}
